// AttentionNet_8710193676417
// MI455X (gfx1250) — compile-verified, matrix operand order checked by reading, not run on hardware
//
#include <hip/hip_runtime.h>

// Problem constants (match reference)
#define BATCH 4096
#define NF    40
#define EMB   64
#define ATT   32
#define NPAIR 780          // NF*(NF-1)/2
#define NPAD  784          // 49 tiles of 16
#define NTILE 49
#define XSTRIDE 68         // EMB + 4 pad -> 16B-aligned rows, conflict-free b128 gathers

typedef __attribute__((ext_vector_type(16))) __bf16 v16bf;
typedef __attribute__((ext_vector_type(8)))  float  v8f;

union AFrag { v16bf v; unsigned int d[8]; };

// Pack two f32 into packed bf16x2 (round-to-nearest) with one v_perm_b32.
static __device__ __forceinline__ unsigned int pack_bf16_rn(float lo, float hi) {
    unsigned int a = __builtin_bit_cast(unsigned int, lo) + 0x8000u;
    unsigned int b = __builtin_bit_cast(unsigned int, hi) + 0x8000u;
    // sel: dst0=a.b2, dst1=a.b3, dst2=b.b2, dst3=b.b3  (0-3 -> src1, 4-7 -> src0)
    return __builtin_amdgcn_perm(b, a, 0x07060302u);
}

// Add lane-pair value via DPP16 (no LDS, no waits). CTRL must be compile-time.
template <int CTRL>
static __device__ __forceinline__ float dpp_add(float x) {
    int moved = __builtin_amdgcn_update_dpp(
        0, __builtin_bit_cast(int, x), CTRL, 0xF, 0xF, true);
    return x + __builtin_bit_cast(float, moved);
}

// Sum across the 16 lanes of each half-wave (rows stay within their half).
static __device__ __forceinline__ float row_reduce16(float t) {
    t = dpp_add<0xB1>(t);   // quad_perm [1,0,3,2]  (xor 1)
    t = dpp_add<0x4E>(t);   // quad_perm [2,3,0,1]  (xor 2)
    t = dpp_add<0x141>(t);  // row_half_mirror      (pairs 4-groups)
    t = dpp_add<0x140>(t);  // row_mirror           (pairs 8-groups)
    return t;
}

__global__ __launch_bounds__(256)
void afm_attention_kernel(const float* __restrict__ x,     // (B, NF, EMB)
                          const float* __restrict__ W,     // (EMB, ATT)
                          const float* __restrict__ bias,  // (ATT)
                          const float* __restrict__ hvec,  // (ATT)
                          const float* __restrict__ pvec,  // (EMB, 1)
                          float* __restrict__ out)         // (B, 1)
{
    __shared__ __align__(16) float xs[(NF + 1) * XSTRIDE]; // staged x[b] + zero row
    __shared__ float ev[NPAD];                  // exp(logit) per pair (padded with 0)
    __shared__ unsigned char pi[NPAIR];         // pair -> i
    __shared__ unsigned char pj[NPAIR];         // pair -> j
    __shared__ float pool[4][EMB];              // pooled partials
    __shared__ float ssum[8];                   // per-wave softmax partials
    __shared__ float total;

    const int tid  = threadIdx.x;
    const int lane = tid & 31;
    const int wave = tid >> 5;
    const int n16  = lane & 15;     // N index / M%16 index inside a tile
    const int h16  = lane >> 4;     // lane half (selects K sub-range / M+8)
    const int b    = blockIdx.x;

    // ---- Stage x[b] into LDS with b128 traffic; build pair tables; zero pads ----
    {
        const float4* xb4 = (const float4*)(x + (size_t)b * (NF * EMB));
        for (int q = tid; q < (NF * EMB) / 4; q += 256) {
            int r  = q >> 4;       // 16 float4 per 64-float row
            int c4 = q & 15;
            *(float4*)(xs + r * XSTRIDE + c4 * 4) = xb4[q];
        }
        if (tid < 16) {            // zero dummy row NF (used by padded lanes)
            float4 z = {0.0f, 0.0f, 0.0f, 0.0f};
            *(float4*)(xs + NF * XSTRIDE + tid * 4) = z;
        }
        for (int p = tid; p < NPAIR; p += 256) {
            int i = 0, c = 0;
            while (c + (NF - 1 - i) <= p) { c += NF - 1 - i; ++i; }
            pi[p] = (unsigned char)i;
            pj[p] = (unsigned char)(i + 1 + (p - c));
        }
        if (tid < NPAD - NPAIR) ev[NPAIR + tid] = 0.0f;
    }

    // ---- Build the four W B-fragments (2 N-tiles x 2 K-steps) per lane ----
    // CDNA5 16-bit fragment layout: lane holds column n16 (half h16);
    // vgpr v, slot s -> k = 16*(v/4) + 8*h16 + 2*(v%4) + s
    AFrag Bf[2][2];
#pragma unroll
    for (int nt = 0; nt < 2; ++nt)
#pragma unroll
    for (int kb = 0; kb < 2; ++kb)
#pragma unroll
    for (int v = 0; v < 8; ++v) {
        int k0 = 16 * (v >> 2) + 8 * h16 + 2 * (v & 3);
        int e0 = kb * 32 + k0;
        int a  = nt * 16 + n16;
        Bf[nt][kb].d[v] = pack_bf16_rn(W[e0 * ATT + a], W[(e0 + 1) * ATT + a]);
    }
    const float bn0 = bias[n16], bn1 = bias[16 + n16];
    const float hn0 = hvec[n16], hn1 = hvec[16 + n16];

    __syncthreads();

    // ---- Attention logits via WMMA: ewp(16x64) @ W(64x32), per M-tile ----
    for (int mt = wave; mt < NTILE; mt += 8) {
        int p = mt * 16 + n16;                 // this lane's pair row
        bool valid = (p < NPAIR);
        int ip = valid ? (int)pi[p] : NF;      // invalid lanes -> zero row
        int jp = valid ? (int)pj[p] : NF;
        const float* xi = xs + ip * XSTRIDE;
        const float* xj = xs + jp * XSTRIDE;

        // Gather this lane's 32 e-values as 4 contiguous 8-float chunks
        // (chunk c covers e = 16*c + 8*h16 + 0..7) using b128 LDS loads.
        float prod[32];
#pragma unroll
        for (int c = 0; c < 4; ++c) {
            int eb = 16 * c + 8 * h16;
            float4 a0 = *(const float4*)(xi + eb);
            float4 a1 = *(const float4*)(xi + eb + 4);
            float4 b0 = *(const float4*)(xj + eb);
            float4 b1 = *(const float4*)(xj + eb + 4);
            prod[8 * c + 0] = a0.x * b0.x;
            prod[8 * c + 1] = a0.y * b0.y;
            prod[8 * c + 2] = a0.z * b0.z;
            prod[8 * c + 3] = a0.w * b0.w;
            prod[8 * c + 4] = a1.x * b1.x;
            prod[8 * c + 5] = a1.y * b1.y;
            prod[8 * c + 6] = a1.z * b1.z;
            prod[8 * c + 7] = a1.w * b1.w;
        }

        // Pack both K-step A fragments: for (kb,v,s),
        // e - 8*h16 = 32*kb + 16*(v>>2) + 2*(v&3) + s  ->  chunk = 2*kb + (v>>2)
        AFrag af0, af1;
#pragma unroll
        for (int v = 0; v < 8; ++v) {
            int w = 2 * (v & 3);
            int c0i = 8 * (v >> 2) + w;
            int c1i = 8 * (2 + (v >> 2)) + w;
            af0.d[v] = pack_bf16_rn(prod[c0i], prod[c0i + 1]);
            af1.d[v] = pack_bf16_rn(prod[c1i], prod[c1i + 1]);
        }

        v8f c0 = {}; v8f c1 = {};
        c0 = __builtin_amdgcn_wmma_f32_16x16x32_bf16(
                 false, af0.v, false, Bf[0][0].v, (short)0, c0, false, false);
        c0 = __builtin_amdgcn_wmma_f32_16x16x32_bf16(
                 false, af1.v, false, Bf[0][1].v, (short)0, c0, false, false);
        c1 = __builtin_amdgcn_wmma_f32_16x16x32_bf16(
                 false, af0.v, false, Bf[1][0].v, (short)0, c1, false, false);
        c1 = __builtin_amdgcn_wmma_f32_16x16x32_bf16(
                 false, af1.v, false, Bf[1][1].v, (short)0, c1, false, false);

        // Epilogue: lane holds N=n16 for rows M = r (+8 for upper half).
        // relu(c+bias)*h summed over both 16-wide N tiles, then DPP 16-lane sum.
#pragma unroll
        for (int r = 0; r < 8; ++r) {
            float a0 = c0[r] + bn0; a0 = a0 > 0.0f ? a0 : 0.0f;
            float a1 = c1[r] + bn1; a1 = a1 > 0.0f ? a1 : 0.0f;
            float t = row_reduce16(a0 * hn0 + a1 * hn1);
            int m = mt * 16 + r + 8 * h16;
            if (n16 == 0 && m < NPAIR) ev[m] = __expf(t);
        }
    }
    __syncthreads();

    // ---- Softmax denominator ----
    {
        float s = 0.0f;
        for (int p = tid; p < NPAD; p += 256) s += ev[p];
        s = row_reduce16(s);
        s += __shfl_xor(s, 16);               // cross-row step
        if (lane == 0) ssum[wave] = s;
    }
    __syncthreads();
    if (tid == 0) {
        float t = 0.0f;
#pragma unroll
        for (int w = 0; w < 8; ++w) t += ssum[w];
        total = t;
    }
    __syncthreads();

    // ---- Attention-weighted pooling: pooled[e] = sum_p ev[p]*ewp[p][e] ----
    {
        int e   = tid & 63;
        int grp = tid >> 6;   // 0..3
        float acc = 0.0f;
        for (int p = grp; p < NPAIR; p += 4) {
            acc += ev[p] * xs[pi[p] * XSTRIDE + e] * xs[pj[p] * XSTRIDE + e];
        }
        pool[grp][e] = acc;
    }
    __syncthreads();

    // ---- Final dot with p (divide by softmax total) ----
    if (tid < EMB) {
        float pe = (pool[0][tid] + pool[1][tid] + pool[2][tid] + pool[3][tid]);
        pool[0][tid] = pe * pvec[tid] / total;
    }
    __syncthreads();
    if (tid == 0) {
        float o = 0.0f;
#pragma unroll
        for (int e = 0; e < EMB; ++e) o += pool[0][e];
        out[b] = o;
    }
}

extern "C" void kernel_launch(void* const* d_in, const int* in_sizes, int n_in,
                              void* d_out, int out_size, void* d_ws, size_t ws_size,
                              hipStream_t stream) {
    (void)in_sizes; (void)n_in; (void)d_ws; (void)ws_size; (void)out_size;
    const float* x    = (const float*)d_in[0];
    const float* W    = (const float*)d_in[1];
    const float* bias = (const float*)d_in[2];
    const float* hvec = (const float*)d_in[3];
    const float* pvec = (const float*)d_in[4];
    float* out = (float*)d_out;

    afm_attention_kernel<<<BATCH, 256, 0, stream>>>(x, W, bias, hvec, pvec, out);
}
